// SequentialRNNV2_43138651521024
// MI455X (gfx1250) — matrix operand order repaired and checked by reading, not compiled, not run on hardware
//
#include <hip/hip_runtime.h>
#include <hip/hip_bf16.h>
#include <stdint.h>
#include <stddef.h>

// ---------------- problem constants ----------------
#define N_SAMP   8192
#define RULE_DIM 32
#define E_DIM    1024
#define H_DIM    1024
#define OUT_DIM  2
#define T_TOT    (N_SAMP * 3)   // 24576 sequential LSTM steps
#define G4H      4096           // 4*H
#define NWG_SEQ  64             // persistent workgroups for the recurrence
#define WHH_STRIDE 1032         // 1024 + 8 pad -> optimal (2-way) banking for b128 reads

typedef __attribute__((ext_vector_type(16))) __bf16 v16bf;
typedef __attribute__((ext_vector_type(8)))  float  v8f;

struct alignas(16) U4 { unsigned a[4]; };

// ---------------- workspace layout ----------------
static constexpr size_t A256(size_t x) { return (x + 255) & ~(size_t)255; }
static constexpr size_t SZ_X  = (size_t)T_TOT * E_DIM * 2;   // bf16 X (one copy)
static constexpr size_t SZ_W  = (size_t)G4H * E_DIM * 2;     // bf16 W_ih (one copy)
static constexpr size_t SZ_G  = (size_t)T_TOT * G4H * 4;     // f32 gate preacts
static constexpr size_t OFF_BAR = 0;                               // 2 x u32
static constexpr size_t OFF_H   = 256;                             // 2 x 1024 f32 (ping-pong h)
static constexpr size_t OFF_XHI = A256(OFF_H + 2 * H_DIM * 4);
static constexpr size_t OFF_XLO = A256(OFF_XHI + SZ_X);
static constexpr size_t OFF_WHI = A256(OFF_XLO + SZ_X);
static constexpr size_t OFF_WLO = A256(OFF_WHI + SZ_W);
static constexpr size_t OFF_G   = A256(OFF_WLO + SZ_W);
static constexpr size_t OFF_END = OFF_G + SZ_G;

__device__ __forceinline__ void bf16_split(float x, __bf16& hi, __bf16& lo) {
  __bf16 h = (__bf16)x;          // round-to-nearest-even
  hi = h;
  lo = (__bf16)(x - (float)h);   // residual
}

__device__ __forceinline__ float sigmoidf_dev(float x) {
  return 1.0f / (1.0f + expf(-x));
}

// ---------------- init: zero barrier + h state ----------------
__global__ void k_init(unsigned* bar, float* hbuf2) {
  int tid = threadIdx.x;
  if (tid < 2) bar[tid] = 0u;
  for (int k = tid; k < 2 * H_DIM; k += blockDim.x) hbuf2[k] = 0.0f;
}

// ---------------- embeddings -> X rows (bf16 hi/lo split) ----------------
__global__ void k_embed(const float* __restrict__ rules,
                        const int*   __restrict__ id1,
                        const int*   __restrict__ id2,
                        const float* __restrict__ W_rule,
                        const float* __restrict__ b_rule,
                        const float* __restrict__ face,
                        __bf16* __restrict__ Xhi, __bf16* __restrict__ Xlo) {
  __shared__ float srow[RULE_DIM];
  int n = blockIdx.x, tid = threadIdx.x;
  if (tid < RULE_DIM) srow[tid] = rules[(size_t)n * RULE_DIM + tid];
  __syncthreads();
  size_t r0 = (size_t)(3 * n) * E_DIM;
  size_t f1 = (size_t)id1[n] * E_DIM;
  size_t f2 = (size_t)id2[n] * E_DIM;
  for (int e = tid; e < E_DIM; e += blockDim.x) {
    float acc = b_rule[e];
#pragma unroll
    for (int k = 0; k < RULE_DIM; ++k) acc += srow[k] * W_rule[(size_t)k * E_DIM + e];
    float rv = 1.0f /*CTX_SCALE*/ * acc;
    __bf16 h, l;
    bf16_split(rv, h, l);                 Xhi[r0 + e] = h;             Xlo[r0 + e] = l;
    bf16_split(face[f1 + e], h, l);       Xhi[r0 + E_DIM + e] = h;     Xlo[r0 + E_DIM + e] = l;
    bf16_split(face[f2 + e], h, l);       Xhi[r0 + 2*E_DIM + e] = h;   Xlo[r0 + 2*E_DIM + e] = l;
  }
}

// ---------------- W_ih -> bf16 hi/lo split ----------------
__global__ void k_wsplit(const float* __restrict__ W_ih,
                         __bf16* __restrict__ Whi, __bf16* __restrict__ Wlo) {
  size_t idx = (size_t)blockIdx.x * blockDim.x + threadIdx.x;
  if (idx >= (size_t)G4H * E_DIM) return;
  __bf16 h, l;
  bf16_split(W_ih[idx], h, l);
  Whi[idx] = h; Wlo[idx] = l;
}

// ---------------- big WMMA GEMM: G = X @ W_ih^T + (b_ih + b_hh) ----------------
// One wave computes a 32x32 output strip (2 M-tiles x 2 N-tiles): A/B register
// reuse halves L2 traffic vs 16x16/wave. 12 WMMAs per K-step (bf16 hi/lo split).
#define MB (T_TOT / 32)   // 768
#define NB (G4H / 32)     // 128
__global__ void k_gemm(const __bf16* __restrict__ Xhi, const __bf16* __restrict__ Xlo,
                       const __bf16* __restrict__ Whi, const __bf16* __restrict__ Wlo,
                       const float* __restrict__ b_ih, const float* __restrict__ b_hh,
                       float* __restrict__ G) {
  int wave = threadIdx.x >> 5;
  int lane = threadIdx.x & 31;
  int gid  = blockIdx.x * 8 + wave;
  int mb = gid % MB;
  int nb = gid / MB;
  size_t t0 = (size_t)mb * 32;
  size_t n0 = (size_t)nb * 32;
  int lm   = lane & 15;
  int hi16 = (lane >> 4) & 1;

  // A (16x32 bf16): m = lane%16 ; slot p holds k pair (p<4?0:16)+(hi16?8:0)+(p&3)*2
  const __bf16* a0hiP = Xhi + (t0 + lm) * E_DIM;
  const __bf16* a0loP = Xlo + (t0 + lm) * E_DIM;
  const __bf16* a1hiP = Xhi + (t0 + 16 + lm) * E_DIM;
  const __bf16* a1loP = Xlo + (t0 + 16 + lm) * E_DIM;
  // B (32x16 bf16): n = lane%16 ; k = hi16*8 + (0..7), then 16 + hi16*8 + (0..7)
  const __bf16* b0hiP = Whi + (n0 + lm) * E_DIM + hi16 * 8;
  const __bf16* b0loP = Wlo + (n0 + lm) * E_DIM + hi16 * 8;
  const __bf16* b1hiP = Whi + (n0 + 16 + lm) * E_DIM + hi16 * 8;
  const __bf16* b1loP = Wlo + (n0 + 16 + lm) * E_DIM + hi16 * 8;

  v8f c00 = {0.f,0.f,0.f,0.f,0.f,0.f,0.f,0.f};
  v8f c01 = c00, c10 = c00, c11 = c00;

  for (int kk = 0; kk < E_DIM; kk += 32) {
    union { v16bf v; unsigned u[8]; } a0h, a0l, a1h, a1l;
    int kb = kk + (hi16 ? 8 : 0);
#pragma unroll
    for (int p = 0; p < 8; ++p) {
      int ko = kb + ((p < 4) ? 0 : 16) + (p & 3) * 2;
      a0h.u[p] = *(const unsigned*)(a0hiP + ko);
      a0l.u[p] = *(const unsigned*)(a0loP + ko);
      a1h.u[p] = *(const unsigned*)(a1hiP + ko);
      a1l.u[p] = *(const unsigned*)(a1loP + ko);
    }
    union { v16bf v; U4 q[2]; } b0h, b0l, b1h, b1l;
    b0h.q[0] = *(const U4*)(b0hiP + kk);  b0h.q[1] = *(const U4*)(b0hiP + kk + 16);
    b0l.q[0] = *(const U4*)(b0loP + kk);  b0l.q[1] = *(const U4*)(b0loP + kk + 16);
    b1h.q[0] = *(const U4*)(b1hiP + kk);  b1h.q[1] = *(const U4*)(b1hiP + kk + 16);
    b1l.q[0] = *(const U4*)(b1loP + kk);  b1l.q[1] = *(const U4*)(b1loP + kk + 16);

    c00 = __builtin_amdgcn_wmma_f32_16x16x32_bf16(false, a0h.v, false, b0h.v, (short)0, c00, false, false);
    c00 = __builtin_amdgcn_wmma_f32_16x16x32_bf16(false, a0h.v, false, b0l.v, (short)0, c00, false, false);
    c00 = __builtin_amdgcn_wmma_f32_16x16x32_bf16(false, a0l.v, false, b0h.v, (short)0, c00, false, false);

    c01 = __builtin_amdgcn_wmma_f32_16x16x32_bf16(false, a0h.v, false, b1h.v, (short)0, c01, false, false);
    c01 = __builtin_amdgcn_wmma_f32_16x16x32_bf16(false, a0h.v, false, b1l.v, (short)0, c01, false, false);
    c01 = __builtin_amdgcn_wmma_f32_16x16x32_bf16(false, a0l.v, false, b1h.v, (short)0, c01, false, false);

    c10 = __builtin_amdgcn_wmma_f32_16x16x32_bf16(false, a1h.v, false, b0h.v, (short)0, c10, false, false);
    c10 = __builtin_amdgcn_wmma_f32_16x16x32_bf16(false, a1h.v, false, b0l.v, (short)0, c10, false, false);
    c10 = __builtin_amdgcn_wmma_f32_16x16x32_bf16(false, a1l.v, false, b0h.v, (short)0, c10, false, false);

    c11 = __builtin_amdgcn_wmma_f32_16x16x32_bf16(false, a1h.v, false, b1h.v, (short)0, c11, false, false);
    c11 = __builtin_amdgcn_wmma_f32_16x16x32_bf16(false, a1h.v, false, b1l.v, (short)0, c11, false, false);
    c11 = __builtin_amdgcn_wmma_f32_16x16x32_bf16(false, a1l.v, false, b1h.v, (short)0, c11, false, false);
  }

  float bias0 = b_ih[n0 + lm]      + b_hh[n0 + lm];
  float bias1 = b_ih[n0 + 16 + lm] + b_hh[n0 + 16 + lm];
#pragma unroll
  for (int r = 0; r < 8; ++r) {
    int m = r + (hi16 ? 8 : 0);   // C/D layout: lanes16-31 hold M=8..15
    G[(t0 + m)      * (size_t)G4H + n0 + lm]      = c00[r] + bias0;
    G[(t0 + m)      * (size_t)G4H + n0 + 16 + lm] = c01[r] + bias1;
    G[(t0 + 16 + m) * (size_t)G4H + n0 + lm]      = c10[r] + bias0;
    G[(t0 + 16 + m) * (size_t)G4H + n0 + 16 + lm] = c11[r] + bias1;
  }
}

// ---------------- device-wide barrier ----------------
__device__ __forceinline__ void grid_barrier(unsigned* bar, int nwg) {
  __threadfence();       // make this thread's global stores visible device-wide
  __syncthreads();
  if (threadIdx.x == 0) {
    unsigned gen  = __hip_atomic_load(bar + 1, __ATOMIC_RELAXED, __HIP_MEMORY_SCOPE_AGENT);
    unsigned prev = __hip_atomic_fetch_add(bar, 1u, __ATOMIC_ACQ_REL, __HIP_MEMORY_SCOPE_AGENT);
    if (prev == (unsigned)(nwg - 1)) {
      __hip_atomic_store(bar, 0u, __ATOMIC_RELAXED, __HIP_MEMORY_SCOPE_AGENT);
      __hip_atomic_fetch_add(bar + 1, 1u, __ATOMIC_RELEASE, __HIP_MEMORY_SCOPE_AGENT);
    } else {
      while (__hip_atomic_load(bar + 1, __ATOMIC_ACQUIRE, __HIP_MEMORY_SCOPE_AGENT) == gen) {
        __builtin_amdgcn_s_sleep(2);
      }
    }
  }
  __syncthreads();
}

// ---------------- persistent sequential LSTM ----------------
// 64 WGs x 256 threads. WG w owns hidden units [16w,16w+16); its 64 W_hh rows
// (i/f/g/o) stay LDS-resident in f32 (~258 KB, using CDNA5's 320KB/WGP LDS).
__global__ void k_seq(const float* __restrict__ W_hh,
                      const float* __restrict__ G,
                      const float* __restrict__ W_out,
                      const float* __restrict__ b_out,
                      float* d_h,               // 2 x 1024 ping-pong (volatile access)
                      unsigned* bar,
                      float* __restrict__ out) {
  extern __shared__ float lds[];
  float* whh  = lds;                          // 64 * WHH_STRIDE
  float* hbuf = lds + 64 * WHH_STRIDE;        // 1024 (16B aligned: 64*1032*4 % 16 == 0)
  float* gbuf = hbuf + H_DIM;                 // 64 gate preacts
  float* cbuf = gbuf + 64;                    // 16 cell states

  int tid = threadIdx.x;
  int wg  = blockIdx.x;

  // stage this WG's W_hh slice (b128 both sides):
  // local row r -> global row (r/16)*1024 + wg*16 + (r%16)
  for (int r = 0; r < 64; ++r) {
    size_t grow = (size_t)(r >> 4) * H_DIM + (size_t)wg * 16 + (r & 15);
    const float4* src = (const float4*)(W_hh + grow * H_DIM);
    float4* dst = (float4*)(whh + r * WHH_STRIDE);
    for (int k4 = tid; k4 < H_DIM / 4; k4 += 256) dst[k4] = src[k4];
  }
  if (tid < 16) cbuf[tid] = 0.0f;
  __syncthreads();

  int wv = tid >> 5, ln = tid & 31;
  int rloc = (wv << 3) + (ln & 7);   // 8 rows per wave
  int seg  = ln >> 3;                // quarter-K owner (float4 chunks k=seg*4+16j)
  const float* wr = whh + (size_t)rloc * WHH_STRIDE;
  size_t growR = (size_t)(rloc >> 4) * H_DIM + (size_t)wg * 16 + (rloc & 15);

  for (int t = 0; t < T_TOT; ++t) {
    int rb = t & 1, wb = rb ^ 1;
    volatile const float* hg = (volatile const float*)(d_h + rb * H_DIM);
    for (int k = tid; k < H_DIM; k += 256) hbuf[k] = hg[k];
    __syncthreads();

    // emit output for the sample that just finished (h carried into this step)
    if (wg == 0 && t > 0 && (t % 3) == 0 && tid < 64) {
      int col = tid >> 5, l2 = tid & 31;
      float a = 0.0f;
      for (int k = l2; k < H_DIM; k += 32) a += hbuf[k] * W_out[(size_t)k * OUT_DIM + col];
      for (int o = 16; o > 0; o >>= 1) a += __shfl_down(a, o);
      if (l2 == 0) out[(size_t)(t / 3 - 1) * OUT_DIM + col] = a + b_out[col];
    }

    // prefetch this row's precomputed x@W_ih^T + bias term
    float gval = (seg == 0) ? G[(size_t)t * G4H + growR] : 0.0f;

    // 64 dot products vs hbuf: float4 LDS reads (ds_load_b128, 2-way banked)
    float acc = 0.0f;
#pragma unroll 4
    for (int j = 0; j < 64; ++j) {
      int k = (seg << 2) + (j << 4);
      float4 w4 = *(const float4*)(wr + k);
      float4 h4 = *(const float4*)(hbuf + k);
      acc += w4.x * h4.x + w4.y * h4.y + w4.z * h4.z + w4.w * h4.w;
    }
    acc += __shfl_down(acc, 16);
    acc += __shfl_down(acc, 8);
    if (seg == 0) gbuf[rloc] = acc + gval;
    __syncthreads();

    if (tid < 16) {
      int u = tid;
      float iv = sigmoidf_dev(gbuf[u]);
      float fv = sigmoidf_dev(gbuf[16 + u]);
      float gv = tanhf(gbuf[32 + u]);
      float ov = sigmoidf_dev(gbuf[48 + u]);
      float cn = fv * cbuf[u] + iv * gv;
      cbuf[u] = cn;
      float hn = ov * tanhf(cn);
      ((volatile float*)d_h)[wb * H_DIM + wg * 16 + u] = hn;
    }
    grid_barrier(bar, NWG_SEQ);
  }

  // final sample's output: last write landed in buffer 0 (T_TOT even)
  if (wg == 0) {
    volatile const float* hg = (volatile const float*)d_h;
    for (int k = tid; k < H_DIM; k += 256) hbuf[k] = hg[k];
    __syncthreads();
    if (tid < 64) {
      int col = tid >> 5, l2 = tid & 31;
      float a = 0.0f;
      for (int k = l2; k < H_DIM; k += 32) a += hbuf[k] * W_out[(size_t)k * OUT_DIM + col];
      for (int o = 16; o > 0; o >>= 1) a += __shfl_down(a, o);
      if (l2 == 0) out[(size_t)(N_SAMP - 1) * OUT_DIM + col] = a + b_out[col];
    }
  }
}

// ---------------- host launch ----------------
extern "C" void kernel_launch(void* const* d_in, const int* in_sizes, int n_in,
                              void* d_out, int out_size, void* d_ws, size_t ws_size,
                              hipStream_t stream) {
  (void)in_sizes; (void)n_in; (void)out_size;
  if (ws_size < OFF_END) return;   // workspace too small; avoid OOB

  const float* rules  = (const float*)d_in[0];
  const int*   id1    = (const int*)  d_in[1];
  const int*   id2    = (const int*)  d_in[2];
  const float* W_rule = (const float*)d_in[3];
  const float* b_rule = (const float*)d_in[4];
  const float* face   = (const float*)d_in[5];
  const float* W_ih   = (const float*)d_in[6];
  const float* W_hh   = (const float*)d_in[7];
  const float* b_ih   = (const float*)d_in[8];
  const float* b_hh   = (const float*)d_in[9];
  const float* W_out  = (const float*)d_in[10];
  const float* b_out  = (const float*)d_in[11];
  float* out = (float*)d_out;

  char* ws = (char*)d_ws;
  unsigned* bar = (unsigned*)(ws + OFF_BAR);
  float*    dh  = (float*)   (ws + OFF_H);
  __bf16*   Xhi = (__bf16*)  (ws + OFF_XHI);
  __bf16*   Xlo = (__bf16*)  (ws + OFF_XLO);
  __bf16*   Whi = (__bf16*)  (ws + OFF_WHI);
  __bf16*   Wlo = (__bf16*)  (ws + OFF_WLO);
  float*    G   = (float*)   (ws + OFF_G);

  static constexpr size_t SEQ_LDS =
      (size_t)(64 * WHH_STRIDE + H_DIM + 64 + 16) * sizeof(float);  // ~262 KB (<320 KB WGP LDS)
  (void)hipFuncSetAttribute((const void*)k_seq,
                            hipFuncAttributeMaxDynamicSharedMemorySize, (int)SEQ_LDS);

  k_init<<<1, 256, 0, stream>>>(bar, dh);
  k_embed<<<N_SAMP, 256, 0, stream>>>(rules, id1, id2, W_rule, b_rule, face, Xhi, Xlo);
  k_wsplit<<<(G4H * E_DIM) / 256, 256, 0, stream>>>(W_ih, Whi, Wlo);
  k_gemm<<<(MB * NB) / 8, 256, 0, stream>>>(Xhi, Xlo, Whi, Wlo, b_ih, b_hh, G);
  k_seq<<<NWG_SEQ, 256, SEQ_LDS, stream>>>(W_hh, G, W_out, b_out, dh, bar, out);
}
